// PositionAwareAttention_35124242547095
// MI455X (gfx1250) — matrix operand order repaired and checked by reading, not compiled, not run on hardware
//
#include <hip/hip_runtime.h>

#define D_MODEL 1024
#define N_HEADS 16
#define HEAD_DIM 64
#define BATCH 2
#define TIME 2048
#define MROWS (BATCH * TIME)   // 4096

typedef __attribute__((ext_vector_type(16))) _Float16 v16h;
typedef __attribute__((ext_vector_type(8)))  _Float16 v8h;
typedef __attribute__((ext_vector_type(8)))  float    v8f;

// ---------------------------------------------------------------------------
// WMMA helper: D = A(16x32 f16) * B(32x16 f16) + C(16x16 f32)
// ---------------------------------------------------------------------------
__device__ __forceinline__ v8f wmma_f16(v16h a, v16h b, v8f c) {
  return __builtin_amdgcn_wmma_f32_16x16x32_f16(
      /*neg_a=*/false, a, /*neg_b=*/false, b,
      /*c_mod=*/(short)0, c, /*reuse_a=*/false, /*reuse_b=*/false);
}

// A fragment (16x32, MxK): lane l holds row m=l%16; with g=l/16,
// halves 0..7 -> K = k0+g*8+0..7, halves 8..15 -> K = k0+g*8+16..23.
__device__ __forceinline__ v16h load_afrag(const _Float16* __restrict__ base,
                                           int ld, int m0, int k0, int lane) {
  int m = m0 + (lane & 15);
  int g = lane >> 4;
  const _Float16* p = base + (size_t)m * ld + k0 + g * 8;
  union { v16h v; v8h h[2]; } u;
  u.h[0] = *(const v8h*)(p);
  u.h[1] = *(const v8h*)(p + 16);
  return u.v;
}

// B fragment (32x16, KxN) with B stored transposed row-major (N x K, ld=ldK):
// lane l holds col n=n0+l%16; with g=l/16, halves 0..7 -> K = k0+g*8+0..7, halves 8..15 -> K = k0+g*8+16..23.
__device__ __forceinline__ v16h load_bfrag(const _Float16* __restrict__ baseT,
                                           int ld, int n0, int k0, int lane) {
  int n = n0 + (lane & 15);
  int g = lane >> 4;
  const _Float16* p = baseT + (size_t)n * ld + k0 + g * 8;
  union { v16h v; v8h h[2]; } u;
  u.h[0] = *(const v8h*)(p);
  u.h[1] = *(const v8h*)(p + 16);
  return u.v;
}

// ---------------------------------------------------------------------------
// f32 -> f16 elementwise (4 elems / thread)
// ---------------------------------------------------------------------------
__global__ __launch_bounds__(256) void cvt_f32_f16_x4(
    const float4* __restrict__ in, _Float16* __restrict__ out) {
  int i = blockIdx.x * 256 + threadIdx.x;
  float4 v = in[i];
  union { _Float16 h[4]; unsigned long long u; } r;
  r.h[0] = (_Float16)v.x; r.h[1] = (_Float16)v.y;
  r.h[2] = (_Float16)v.z; r.h[3] = (_Float16)v.w;
  *(unsigned long long*)(out + (size_t)i * 4) = r.u;
}

// ---------------------------------------------------------------------------
// Transpose + convert 1024x1024 f32 W -> f16 Wt (Wt[n][k] = W[k][n])
// ---------------------------------------------------------------------------
__global__ __launch_bounds__(256) void transpose_cvt(
    const float* __restrict__ W, _Float16* __restrict__ Wt) {
  __shared__ _Float16 tile[32][33];
  int bx = blockIdx.x * 32;   // col of W
  int by = blockIdx.y * 32;   // row of W
  int tx = threadIdx.x & 31;
  int ty = threadIdx.x >> 5;  // 0..7
#pragma unroll
  for (int i = 0; i < 32; i += 8)
    tile[ty + i][tx] = (_Float16)W[(size_t)(by + ty + i) * 1024 + bx + tx];
  __syncthreads();
#pragma unroll
  for (int i = 0; i < 32; i += 8)
    Wt[(size_t)(bx + ty + i) * 1024 + by + tx] = tile[tx][ty + i];
}

// ---------------------------------------------------------------------------
// Limb-darkening intensity: normalized intensity In[b][t] (f32)
// ---------------------------------------------------------------------------
__global__ __launch_bounds__(256) void intensity_kernel(
    const float* __restrict__ sp, float* __restrict__ In) {
  int b = blockIdx.x;
  float u1 = fminf(fmaxf(sp[b * 7 + 5], 0.f), 1.f);
  float u2 = fminf(fmaxf(sp[b * 7 + 6], 0.f), 1.f);
  float bi = fminf(fmaxf(sp[b * 7 + 4], 0.f), 0.95f);
  float b2 = bi * bi;
  __shared__ float red[8];
  float s = 0.f;
  for (int t = threadIdx.x; t < TIME; t += 256) {
    float pos = -1.f + 2.f * (float)t / (float)(TIME - 1);
    float sv  = fminf(fabsf(pos), 1.f);
    float r2  = fminf(fmaxf(b2 + (1.f - b2) * sv * sv, 0.f), 0.999999f);
    float mu  = sqrtf(fmaxf(1.f - r2, 0.f));
    float omm = 1.f - mu;
    float I   = 1.f - u1 * omm - u2 * omm * omm;
    In[b * TIME + t] = I;
    s += I;
  }
#pragma unroll
  for (int off = 1; off < 32; off <<= 1) s += __shfl_xor(s, off, 32);
  if ((threadIdx.x & 31) == 0) red[threadIdx.x >> 5] = s;
  __syncthreads();
  float tot = 0.f;
#pragma unroll
  for (int w = 0; w < 8; w++) tot += red[w];
  float mean = fmaxf(tot / (float)TIME, 1e-8f);
  for (int t = threadIdx.x; t < TIME; t += 256) In[b * TIME + t] /= mean;
}

// ---------------------------------------------------------------------------
// QKV projection GEMM: out = X(4096x1024,f16) @ Wt^T + bias, stored f16.
// layout 0: out[((b*H+h)*T + t)*64 + d]   (Q, K)
// layout 1: out[((b*H+h)*64 + d)*T + t]   (V transposed, for PV B-operand)
// Block: 256 thr (8 waves), tile 128x64, wave = 32x32 (2x2 WMMA frags).
// ---------------------------------------------------------------------------
__global__ __launch_bounds__(256) void gemm_qkv_kernel(
    const _Float16* __restrict__ X, const _Float16* __restrict__ Wt,
    const float* __restrict__ bias, _Float16* __restrict__ out, int layout) {
  const int K = 1024;
  int wid  = threadIdx.x >> 5;
  int lane = threadIdx.x & 31;
  int m0 = blockIdx.x * 128 + (wid >> 1) * 32;
  int n0 = blockIdx.y * 64  + (wid & 1)  * 32;
  v8f acc[2][2] = {};
  for (int k0 = 0; k0 < K; k0 += 32) {
    if (k0 + 128 < K) {
      __builtin_prefetch(X  + (size_t)(m0 + (lane & 15)) * K + k0 + 128, 0, 1);
      __builtin_prefetch(Wt + (size_t)(n0 + (lane & 15)) * K + k0 + 128, 0, 1);
    }
    v16h a0 = load_afrag(X, K, m0,      k0, lane);
    v16h a1 = load_afrag(X, K, m0 + 16, k0, lane);
    v16h b0 = load_bfrag(Wt, K, n0,      k0, lane);
    v16h b1 = load_bfrag(Wt, K, n0 + 16, k0, lane);
    acc[0][0] = wmma_f16(a0, b0, acc[0][0]);
    acc[0][1] = wmma_f16(a0, b1, acc[0][1]);
    acc[1][0] = wmma_f16(a1, b0, acc[1][0]);
    acc[1][1] = wmma_f16(a1, b1, acc[1][1]);
  }
  int g8 = (lane >> 4) * 8;
#pragma unroll
  for (int i = 0; i < 2; i++) {
#pragma unroll
    for (int j = 0; j < 2; j++) {
      int n  = n0 + j * 16 + (lane & 15);
      float bv = bias[n];
      int h = n >> 6, d = n & 63;
#pragma unroll
      for (int r = 0; r < 8; r++) {
        int m = m0 + i * 16 + r + g8;
        int b = m >> 11, t = m & (TIME - 1);
        float val = acc[i][j][r] + bv;
        size_t addr = (layout == 0)
            ? ((size_t)((b * N_HEADS + h) * TIME + t)) * 64 + d
            : ((size_t)((b * N_HEADS + h) * 64 + d)) * TIME + t;
        out[addr] = (_Float16)val;
      }
    }
  }
}

// ---------------------------------------------------------------------------
// Flash attention. Grid: (B*H, T/64). Block 128 thr = 4 waves.
// Each wave: 16 query rows, full pass over 2048 keys in 32-key steps.
// Pair bias reconstructed as log(clip(In[q]*In[k], eps)).
// ---------------------------------------------------------------------------
__global__ __launch_bounds__(128) void attn_flash_kernel(
    const _Float16* __restrict__ Qb, const _Float16* __restrict__ Kb,
    const _Float16* __restrict__ Vt, const float* __restrict__ In,
    _Float16* __restrict__ att) {
  __shared__ __align__(16) _Float16 plds[4][16 * 32];  // per-wave P tile
  int wid  = threadIdx.x >> 5;
  int lane = threadIdx.x & 31;
  int bh = blockIdx.x;
  int b  = bh >> 4;
  int h  = bh & 15;
  int q0 = blockIdx.y * 64 + wid * 16;
  const _Float16* Qh = Qb + (size_t)bh * TIME * 64;
  const _Float16* Kh = Kb + (size_t)bh * TIME * 64;
  const _Float16* Vh = Vt + (size_t)bh * 64 * TIME;   // (64 x T) row-major
  const float* Ib = In + b * TIME;
  const float scale = 0.125f;  // 64^-0.5
  _Float16* myl = plds[wid];
  int g8 = (lane >> 4) * 8;
  int nn = lane & 15;

  v16h qa0 = load_afrag(Qh, 64, q0, 0,  lane);
  v16h qa1 = load_afrag(Qh, 64, q0, 32, lane);

  float rI[8], mrow[8], lrow[8];
#pragma unroll
  for (int r = 0; r < 8; r++) {
    rI[r]   = Ib[q0 + r + g8];
    mrow[r] = -1e30f;
    lrow[r] = 0.f;
  }
  v8f o0 = {}, o1 = {}, o2 = {}, o3 = {};

  for (int kt = 0; kt < TIME; kt += 32) {
    // S = Q K^T for 32 keys (two 16-key fragments, K-dim = head_dim 64)
    v16h kb00 = load_bfrag(Kh, 64, kt,      0,  lane);
    v16h kb01 = load_bfrag(Kh, 64, kt,      32, lane);
    v16h kb10 = load_bfrag(Kh, 64, kt + 16, 0,  lane);
    v16h kb11 = load_bfrag(Kh, 64, kt + 16, 32, lane);
    v8f s0 = {}, s1 = {};
    s0 = wmma_f16(qa0, kb00, s0);
    s0 = wmma_f16(qa1, kb01, s0);
    s1 = wmma_f16(qa0, kb10, s1);
    s1 = wmma_f16(qa1, kb11, s1);

    float Ik0 = Ib[kt + nn];
    float Ik1 = Ib[kt + 16 + nn];
    float p0[8], p1[8];
#pragma unroll
    for (int r = 0; r < 8; r++) {
      p0[r] = s0[r] * scale + logf(fmaxf(rI[r] * Ik0, 1e-8f));
      p1[r] = s1[r] * scale + logf(fmaxf(rI[r] * Ik1, 1e-8f));
    }
    // online softmax per row (rows live across 16-lane half-groups)
#pragma unroll
    for (int r = 0; r < 8; r++) {
      float mx = fmaxf(p0[r], p1[r]);
#pragma unroll
      for (int off = 1; off < 16; off <<= 1) mx = fmaxf(mx, __shfl_xor(mx, off, 32));
      float nm = fmaxf(mrow[r], mx);
      float ef = __expf(mrow[r] - nm);
      mrow[r] = nm;
      p0[r] = __expf(p0[r] - nm);
      p1[r] = __expf(p1[r] - nm);
      float rs = p0[r] + p1[r];
#pragma unroll
      for (int off = 1; off < 16; off <<= 1) rs += __shfl_xor(rs, off, 32);
      lrow[r] = lrow[r] * ef + rs;
      o0[r] *= ef; o1[r] *= ef; o2[r] *= ef; o3[r] *= ef;
    }
    // C-layout -> A-layout via per-wave LDS tile (same-wave LDS is in-order)
#pragma unroll
    for (int r = 0; r < 8; r++) {
      int m = r + g8;
      myl[m * 32 + nn]      = (_Float16)p0[r];
      myl[m * 32 + 16 + nn] = (_Float16)p1[r];
    }
    v16h pa;
    {
      int m = lane & 15;
      int g = lane >> 4;
      union { v16h v; v8h h2[2]; } u;
      u.h2[0] = *(const v8h*)(myl + m * 32 + g * 8);
      u.h2[1] = *(const v8h*)(myl + m * 32 + g * 8 + 16);
      pa = u.v;
    }
    // O += P * V  (V stored transposed: lane col d, contiguous keys)
    v16h vb0 = load_bfrag(Vh, TIME, 0,  kt, lane);
    v16h vb1 = load_bfrag(Vh, TIME, 16, kt, lane);
    v16h vb2 = load_bfrag(Vh, TIME, 32, kt, lane);
    v16h vb3 = load_bfrag(Vh, TIME, 48, kt, lane);
    o0 = wmma_f16(pa, vb0, o0);
    o1 = wmma_f16(pa, vb1, o1);
    o2 = wmma_f16(pa, vb2, o2);
    o3 = wmma_f16(pa, vb3, o3);
  }
  // normalize + write attended (B,T,D) f16
#pragma unroll
  for (int r = 0; r < 8; r++) {
    float inv = 1.f / lrow[r];
    int m = r + g8;
    int t = q0 + m;
    size_t rowbase = ((size_t)(b * TIME + t)) * D_MODEL + h * 64 + nn;
    att[rowbase +  0] = (_Float16)(o0[r] * inv);
    att[rowbase + 16] = (_Float16)(o1[r] * inv);
    att[rowbase + 32] = (_Float16)(o2[r] * inv);
    att[rowbase + 48] = (_Float16)(o3[r] * inv);
  }
}

// ---------------------------------------------------------------------------
// Output projection + residual: y = att @ Wo + bo + x  (f32)
// ---------------------------------------------------------------------------
__global__ __launch_bounds__(256) void gemm_out_kernel(
    const _Float16* __restrict__ A, const _Float16* __restrict__ Wt,
    const float* __restrict__ bias, const float* __restrict__ x,
    float* __restrict__ y) {
  const int K = 1024;
  int wid  = threadIdx.x >> 5;
  int lane = threadIdx.x & 31;
  int m0 = blockIdx.x * 128 + (wid >> 1) * 32;
  int n0 = blockIdx.y * 64  + (wid & 1)  * 32;
  v8f acc[2][2] = {};
  for (int k0 = 0; k0 < K; k0 += 32) {
    if (k0 + 128 < K) {
      __builtin_prefetch(A  + (size_t)(m0 + (lane & 15)) * K + k0 + 128, 0, 1);
      __builtin_prefetch(Wt + (size_t)(n0 + (lane & 15)) * K + k0 + 128, 0, 1);
    }
    v16h a0 = load_afrag(A, K, m0,      k0, lane);
    v16h a1 = load_afrag(A, K, m0 + 16, k0, lane);
    v16h b0 = load_bfrag(Wt, K, n0,      k0, lane);
    v16h b1 = load_bfrag(Wt, K, n0 + 16, k0, lane);
    acc[0][0] = wmma_f16(a0, b0, acc[0][0]);
    acc[0][1] = wmma_f16(a0, b1, acc[0][1]);
    acc[1][0] = wmma_f16(a1, b0, acc[1][0]);
    acc[1][1] = wmma_f16(a1, b1, acc[1][1]);
  }
  int g8 = (lane >> 4) * 8;
#pragma unroll
  for (int i = 0; i < 2; i++) {
#pragma unroll
    for (int j = 0; j < 2; j++) {
      int n = n0 + j * 16 + (lane & 15);
      float bv = bias[n];
#pragma unroll
      for (int r = 0; r < 8; r++) {
        int m = m0 + i * 16 + r + g8;
        size_t idx = (size_t)m * D_MODEL + n;
        y[idx] = acc[i][j][r] + bv + x[idx];
      }
    }
  }
}

// ---------------------------------------------------------------------------
// Row LayerNorm: one block per row of 1024
// ---------------------------------------------------------------------------
__global__ __launch_bounds__(256) void layernorm_kernel(
    const float* __restrict__ y, const float* __restrict__ gamma,
    const float* __restrict__ beta, float* __restrict__ out) {
  __shared__ float rs[8], rs2[8];
  int row = blockIdx.x;
  const float* yr = y + (size_t)row * D_MODEL;
  float s = 0.f, s2 = 0.f;
  for (int i = threadIdx.x; i < D_MODEL; i += 256) {
    float v = yr[i];
    s += v; s2 += v * v;
  }
#pragma unroll
  for (int off = 1; off < 32; off <<= 1) {
    s  += __shfl_xor(s,  off, 32);
    s2 += __shfl_xor(s2, off, 32);
  }
  if ((threadIdx.x & 31) == 0) { rs[threadIdx.x >> 5] = s; rs2[threadIdx.x >> 5] = s2; }
  __syncthreads();
  float ts = 0.f, ts2 = 0.f;
#pragma unroll
  for (int w = 0; w < 8; w++) { ts += rs[w]; ts2 += rs2[w]; }
  float mean = ts * (1.f / D_MODEL);
  float var  = ts2 * (1.f / D_MODEL) - mean * mean;
  float rstd = rsqrtf(var + 1e-5f);
  for (int i = threadIdx.x; i < D_MODEL; i += 256)
    out[(size_t)row * D_MODEL + i] = (yr[i] - mean) * rstd * gamma[i] + beta[i];
}

// ---------------------------------------------------------------------------
extern "C" void kernel_launch(void* const* d_in, const int* in_sizes, int n_in,
                              void* d_out, int out_size, void* d_ws, size_t ws_size,
                              hipStream_t stream) {
  const float* x    = (const float*)d_in[0];
  const float* sp   = (const float*)d_in[1];
  const float* Wq   = (const float*)d_in[2];
  const float* bq   = (const float*)d_in[3];
  const float* Wk   = (const float*)d_in[4];
  const float* bk   = (const float*)d_in[5];
  const float* Wv   = (const float*)d_in[6];
  const float* bv   = (const float*)d_in[7];
  const float* Wo   = (const float*)d_in[8];
  const float* bo   = (const float*)d_in[9];
  const float* gamma= (const float*)d_in[10];
  const float* beta = (const float*)d_in[11];
  float* out = (float*)d_out;

  // workspace carve-out (~64 MB total)
  char* ws = (char*)d_ws;
  size_t off = 0;
  auto carve = [&](size_t bytes) { void* p = ws + off; off += (bytes + 255) & ~(size_t)255; return p; };
  _Float16* xh  = (_Float16*)carve((size_t)MROWS * D_MODEL * 2);
  _Float16* Wqt = (_Float16*)carve((size_t)D_MODEL * D_MODEL * 2);
  _Float16* Wkt = (_Float16*)carve((size_t)D_MODEL * D_MODEL * 2);
  _Float16* Wvt = (_Float16*)carve((size_t)D_MODEL * D_MODEL * 2);
  _Float16* Wot = (_Float16*)carve((size_t)D_MODEL * D_MODEL * 2);
  _Float16* Qb  = (_Float16*)carve((size_t)MROWS * D_MODEL * 2);
  _Float16* Kbf = (_Float16*)carve((size_t)MROWS * D_MODEL * 2);
  _Float16* Vtb = (_Float16*)carve((size_t)MROWS * D_MODEL * 2);
  _Float16* att = (_Float16*)carve((size_t)MROWS * D_MODEL * 2);
  float*    In  = (float*)carve((size_t)BATCH * TIME * 4);
  float*    yb  = (float*)carve((size_t)MROWS * D_MODEL * 4);

  // 1) conversions
  cvt_f32_f16_x4<<<(MROWS * D_MODEL) / (256 * 4), 256, 0, stream>>>((const float4*)x, xh);
  dim3 tg(32, 32);
  transpose_cvt<<<tg, 256, 0, stream>>>(Wq, Wqt);
  transpose_cvt<<<tg, 256, 0, stream>>>(Wk, Wkt);
  transpose_cvt<<<tg, 256, 0, stream>>>(Wv, Wvt);
  transpose_cvt<<<tg, 256, 0, stream>>>(Wo, Wot);
  intensity_kernel<<<BATCH, 256, 0, stream>>>(sp, In);

  // 2) QKV projections (WMMA)
  dim3 gg(MROWS / 128, D_MODEL / 64);
  gemm_qkv_kernel<<<gg, 256, 0, stream>>>(xh, Wqt, bq, Qb,  0);
  gemm_qkv_kernel<<<gg, 256, 0, stream>>>(xh, Wkt, bk, Kbf, 0);
  gemm_qkv_kernel<<<gg, 256, 0, stream>>>(xh, Wvt, bv, Vtb, 1);  // V transposed

  // 3) flash attention with pair bias (WMMA)
  attn_flash_kernel<<<dim3(BATCH * N_HEADS, TIME / 64), 128, 0, stream>>>(Qb, Kbf, Vtb, In, att);

  // 4) output projection + residual (WMMA), then LayerNorm
  gemm_out_kernel<<<gg, 256, 0, stream>>>(att, Wot, bo, x, yb);
  layernorm_kernel<<<MROWS, 256, 0, stream>>>(yb, gamma, beta, out);
}
